// TemporalSSMPatchwise_2173253452151
// MI455X (gfx1250) — matrix operand order repaired and checked by reading, not compiled, not run on hardware
//
#include <hip/hip_runtime.h>

typedef __bf16 bf16_t;
typedef __bf16 v8bf  __attribute__((ext_vector_type(8)));
typedef __bf16 v16bf __attribute__((ext_vector_type(16)));
typedef float  v8f   __attribute__((ext_vector_type(8)));
typedef unsigned int u32x4 __attribute__((ext_vector_type(4)));
typedef int          i32x4 __attribute__((ext_vector_type(4)));
typedef int          i32x8 __attribute__((ext_vector_type(8)));

#define NTHR 128
#define NBLK 2400
// LDS weight block element offsets (bf16 elements)
#define OFF_WE   0
#define OFF_BB   4096
#define OFF_CC   8192
#define OFF_WENC 12288
#define OFF_WDEC 20480
#define W_ELEMS  24576   // 49152 bytes total staged weights (12288 dwords)
// workspace byte offsets
#define WS_WB    0
#define WS_LBAR  49152
#define WS_WOUT  49664

#if defined(__gfx1250__) && __has_builtin(__builtin_amdgcn_tensor_load_to_lds) && \
    __has_builtin(__builtin_amdgcn_s_wait_tensorcnt)
#define HAVE_TDM 1
#else
#define HAVE_TDM 0
#endif

__device__ __forceinline__ bf16_t f2bf(float f) {
  union { float f; unsigned u; } v; v.f = f;
  unsigned r = v.u + 0x7FFFu + ((v.u >> 16) & 1u);
  unsigned short h = (unsigned short)(r >> 16);
  return __builtin_bit_cast(bf16_t, h);
}

__device__ __forceinline__ float gelu_exact(float x) {
  return 0.5f * x * (1.0f + erff(x * 0.70710678118654752f));
}

#if HAVE_TDM
// LDS byte offset of a __shared__ object (AS(3) pointer value == LDS offset)
__device__ __forceinline__ unsigned lds_off(const void* p) {
  return (unsigned)(unsigned long long)(__attribute__((address_space(3))) const void*)p;
}
// Issue a 2-D TDM tile load (dword elements) per CDNA5 ISA D# layout (§8).
// tile: tile_d0 x tile_d1 dwords; row stride in tensor = stride0 dwords.
__device__ __forceinline__ void tdm_load_2d(unsigned lds, const void* gaddr,
                                            unsigned tile_d0, unsigned tile_d1,
                                            unsigned tens_d0, unsigned tens_d1,
                                            unsigned stride0) {
  unsigned long long ga = (unsigned long long)gaddr;
  // group0: [1:0]=count=1 | [63:32]=lds_addr | [120:64]=global_addr | [127:126]=type(2)
  u32x4 g0 = { 1u, lds, (unsigned)ga,
               (unsigned)((ga >> 32) & 0x01FFFFFFu) | (2u << 30) };
  i32x8 g1;
  g1[0] = 0x00020000;                                        // data_size = 4B
  g1[1] = (int)((tens_d0 & 0xFFFFu) << 16);                  // tensor_dim0[15:0] @63:48
  g1[2] = (int)((tens_d0 >> 16) | ((tens_d1 & 0xFFFFu) << 16)); // d0 hi, tensor_dim1 lo
  g1[3] = (int)((tens_d1 >> 16) | (tile_d0 << 16));          // tensor_dim1 hi, tile_dim0
  g1[4] = (int)tile_d1;                                      // tile_dim1 (tile_dim2=0)
  g1[5] = (int)stride0;                                      // tensor_dim0_stride[31:0]
  g1[6] = 0;                                                 // stride0 hi, stride1 lo
  g1[7] = 0;
  i32x4 g2 = { 1, 0, 0, 0 };                                 // tensor_dim2=1
  i32x4 g3 = { 0, 0, 0, 0 };
  i32x8 g4 = { 0, 0, 0, 0, 0, 0, 0, 0 };                     // extra operand (zero-filled)
  __builtin_amdgcn_tensor_load_to_lds(g0, g1, g2, g3, g4, 0);
}
#endif

// ---- WMMA fragment loaders (CDNA5 16-bit layouts, wave32) ----
// A (16x32 bf16): lanes 0-15: M=lane, halfs = K[0..7] then K[16..23];
//                 lanes 16-31: M=lane-16, halfs = K[8..15] then K[24..31].
__device__ __forceinline__ v16bf ldA(const bf16_t* A, int kb, int lane) {
  int m  = lane & 15;
  int kh = (lane >> 4) << 3;
  int base = m * 64 + kb * 32 + kh;
  v8bf lo = *(const v8bf*)(A + base);
  v8bf hi = *(const v8bf*)(A + base + 16);
  return __builtin_shufflevector(lo, hi, 0,1,2,3,4,5,6,7,8,9,10,11,12,13,14,15);
}
// A-fragment from f32 input tile in TDM row order: element (s, k=i*8+j) at [i*128 + s*8 + j]
__device__ __forceinline__ v16bf ldA_xin(const float* X, int kb, int lane) {
  int m  = lane & 15;
  int kh = (lane >> 4) << 3;
  int k0 = kb * 32 + kh;       // multiple of 8
  int k1 = k0 + 16;
  const float* p0 = X + (k0 >> 3) * 128 + m * 8;
  const float* p1 = X + (k1 >> 3) * 128 + m * 8;
  v16bf r;
  #pragma unroll
  for (int e = 0; e < 8; ++e) r[e] = f2bf(p0[e]);
  #pragma unroll
  for (int e = 0; e < 8; ++e) r[8 + e] = f2bf(p1[e]);
  return r;
}
// B (32x16 bf16) from row-major Wt[N][64] (Wt[n][k] == B[k][n]):
// lanes 0-15: N=lane, K=0..7 then 16..23; lanes 16-31: N=lane-16, K=8..15 then 24..31.
__device__ __forceinline__ v16bf ldB(const bf16_t* Wt, int n0, int kb, int lane) {
  int n  = lane & 15;
  int ks = (lane >> 4) << 3;
  int base = (n0 + n) * 64 + kb * 32 + ks;
  v8bf lo = *(const v8bf*)(Wt + base);
  v8bf hi = *(const v8bf*)(Wt + base + 16);
  return __builtin_shufflevector(lo, hi, 0,1,2,3,4,5,6,7,8,9,10,11,12,13,14,15);
}
// full K=64 GEMM tile: D(16x16) = A(16x64) * Wt^T(64x16), f32 accumulate
__device__ __forceinline__ v8f wmma64(const bf16_t* A, const bf16_t* Wt, int n0, int lane) {
  v8f acc = {};
  acc = __builtin_amdgcn_wmma_f32_16x16x32_bf16(false, ldA(A, 0, lane),
                                                false, ldB(Wt, n0, 0, lane),
                                                (short)0, acc, false, false);
  acc = __builtin_amdgcn_wmma_f32_16x16x32_bf16(false, ldA(A, 1, lane),
                                                false, ldB(Wt, n0, 1, lane),
                                                (short)0, acc, false, false);
  return acc;
}
__device__ __forceinline__ v8f wmma64_xin(const float* X, const bf16_t* Wt, int n0, int lane) {
  v8f acc = {};
  acc = __builtin_amdgcn_wmma_f32_16x16x32_bf16(false, ldA_xin(X, 0, lane),
                                                false, ldB(Wt, n0, 0, lane),
                                                (short)0, acc, false, false);
  acc = __builtin_amdgcn_wmma_f32_16x16x32_bf16(false, ldA_xin(X, 1, lane),
                                                false, ldB(Wt, n0, 1, lane),
                                                (short)0, acc, false, false);
  return acc;
}

// LayerNorm over 16 rows x 64 cols (f32 LDS in) -> bf16 LDS out (+optional f32 out)
__device__ __forceinline__ void layernorm_16x64(const float* in,
                                                const float* __restrict__ g,
                                                const float* __restrict__ b,
                                                bf16_t* outb, float* outf,
                                                float* sMu, float* sRs, int tid) {
  if (tid < 16) {
    const float* row = in + tid * 64;
    float s = 0.f;
    #pragma unroll 8
    for (int k = 0; k < 64; ++k) s += row[k];
    float mu = s * (1.f / 64.f);
    float var = 0.f;
    #pragma unroll 8
    for (int k = 0; k < 64; ++k) { float d = row[k] - mu; var += d * d; }
    sMu[tid] = mu;
    sRs[tid] = rsqrtf(var * (1.f / 64.f) + 1e-5f);
  }
  __syncthreads();
  #pragma unroll
  for (int e = tid; e < 1024; e += NTHR) {
    int r = e >> 6, k = e & 63;
    float v = (in[e] - sMu[r]) * sRs[r] * g[k] + b[k];
    outb[e] = f2bf(v);
    if (outf) outf[e] = v;
  }
  __syncthreads();
}

// ---------------- prep: fold params into bf16 weight blocks ----------------
__global__ __launch_bounds__(256)
void ssm_prep(const float* __restrict__ W_embed,
              const float* __restrict__ Lre, const float* __restrict__ Lim,
              const float* __restrict__ B_re, const float* __restrict__ B_im,
              const float* __restrict__ C_re, const float* __restrict__ C_im,
              const float* __restrict__ log_step,
              const float* __restrict__ W_enc, const float* __restrict__ W_dec,
              const float* __restrict__ W_out,
              bf16_t* __restrict__ wWb, float* __restrict__ wLbar,
              bf16_t* __restrict__ wWout) {
  int i0 = blockIdx.x * blockDim.x + threadIdx.x;
  int stride = gridDim.x * blockDim.x;
  for (int i = i0; i < 4096; i += stride) wWb[OFF_WE + i] = f2bf(W_embed[i]);
  // Bbar^T block: cols 0..31 = re, 32..63 = im ; Bbar = ((Lbar-1)/Lam) * (B_re + i B_im)
  for (int i = i0; i < 4096; i += stride) {
    int col = i >> 6, k = i & 63, p = col & 31;
    float st = __expf(log_step[p]);
    float lr = Lre[p], li = Lim[p];
    float er  = __expf(lr * st);
    float Lbr = er * __cosf(li * st);
    float Lbi = er * __sinf(li * st);
    float cr = Lbr - 1.f, ci = Lbi;
    float inv = 1.f / (lr * lr + li * li);
    float qr = (cr * lr + ci * li) * inv;
    float qi = (ci * lr - cr * li) * inv;
    float br = B_re[p * 64 + k], bi = B_im[p * 64 + k];
    float val = (col < 32) ? (qr * br - qi * bi) : (qr * bi + qi * br);
    wWb[OFF_BB + i] = f2bf(val);
  }
  // Ccat^T block with bandlimit mask
  for (int i = i0; i < 4096; i += stride) {
    int d = i >> 6, pc = i & 63, p = pc & 31;
    float st = __expf(log_step[p]);
    float freq = st * fabsf(Lim[p]) * 0.15915494309189535f;
    float mask = (freq < 0.5f) ? 1.f : 0.f;
    float val = (pc < 32) ? (C_re[d * 32 + p] * mask) : (-C_im[d * 32 + p] * mask);
    wWb[OFF_CC + i] = f2bf(val);
  }
  for (int i = i0; i < 8192; i += stride) wWb[OFF_WENC + i] = f2bf(W_enc[i]);
  for (int i = i0; i < 4096; i += stride) wWb[OFF_WDEC + i] = f2bf(W_dec[i]);
  for (int i = i0; i < 64; i += stride) {
    int p = i & 31;
    float st = __expf(log_step[p]);
    float er = __expf(Lre[p] * st);
    wLbar[i] = (i < 32) ? er * __cosf(Lim[p] * st) : er * __sinf(Lim[p] * st);
  }
  for (int i = i0; i < 65536; i += stride) wWout[i] = f2bf(W_out[i]);
}

// ---------------- fused main kernel: 1 WG = 16 sequences ----------------
__global__ __launch_bounds__(NTHR)
void ssm_fused(const float* __restrict__ x,
               const float* __restrict__ b_embed,
               const float* __restrict__ g1, const float* __restrict__ be1,
               const float* __restrict__ Dv,
               const float* __restrict__ g2, const float* __restrict__ be2,
               const float* __restrict__ g3, const float* __restrict__ be3,
               const float* __restrict__ b_out,
               const bf16_t* __restrict__ wWb,
               const float*  __restrict__ wLbar,
               const bf16_t* __restrict__ wWout,
               float* __restrict__ y) {
  __shared__ __attribute__((aligned(16))) bf16_t sW[W_ELEMS];     // staged weights (48KB)
  __shared__ __attribute__((aligned(16))) float  sXin[16 * 64];   // f32 input tile (TDM order)
  __shared__ __attribute__((aligned(16))) bf16_t sA[16 * 64];     // bf16 A staging (xs / h)
  __shared__ __attribute__((aligned(16))) bf16_t sFxB[16 * 64];   // bf16 fx / fx2 / xlast
  __shared__ __attribute__((aligned(16))) float  sX[16 * 64];     // f32 staging (xe/xb/xl)
  __shared__ __attribute__((aligned(16))) float  sBu[16 * 64];    // Bu_t (re|im)
  __shared__ __attribute__((aligned(16))) float  sFx14[16 * 64];  // fx at t=14
  __shared__ __attribute__((aligned(16))) float  sFx2f[16 * 64];  // fx2 f32 residual
  __shared__ __attribute__((aligned(16))) float  sEnc[16 * 128];  // enc output
  __shared__ __attribute__((aligned(16))) float  sOutT[16 * 64];  // output staging
  __shared__ float sMu[16], sRs[16];

  const int tid  = threadIdx.x;
  const int wave = tid >> 5;
  const int lane = tid & 31;

  const int g    = blockIdx.x;         // 0..2399
  const int bImg = g / 300;
  const int pg   = g % 300;
  const int py   = pg / 5;             // patch row (0..59)
  const int px0  = (pg % 5) * 16;      // first patch col in this group

  // ---- stage weights to LDS (TDM: one 49152-byte tile; fallback: float4 copy) ----
#if HAVE_TDM
  if (wave == 0) {
    tdm_load_2d(lds_off(sW), wWb, /*tile*/ 12288u, 1u, /*tensor*/ 12288u, 1u, 12288u);
    __builtin_amdgcn_s_wait_tensorcnt(0);
  }
#else
  {
    const float4* src = (const float4*)wWb;
    float4* dst = (float4*)sW;
    #pragma unroll
    for (int c = tid; c < 3072; c += NTHR) dst[c] = src[c];
  }
#endif
  // warm W_out (bf16, 128KB -> L2 resident across all 2400 WGs)
  #pragma unroll
  for (int c = tid; c < 1024; c += NTHR)
    __builtin_prefetch(wWout + c * 64, 0, 1);

  // scan state in registers: 4 complex states per thread
  float l_re[4], l_im[4], s_re[4], s_im[4];
  #pragma unroll
  for (int q = 0; q < 4; ++q) {
    int p = (tid * 4 + q) & 31;
    l_re[q] = wLbar[p];
    l_im[q] = wLbar[32 + p];
    s_re[q] = 0.f;
    s_im[q] = 0.f;
  }
  __syncthreads();

  const int s_  = tid >> 3;   // sequence 0..15 for 128-wide coalesced I/O
  const int j_  = tid & 7;    // col-in-patch
  const int n   = lane & 15;  // D-frag column within tile
  const int mo  = (lane >> 4) << 3;

  // ---------- temporal loop ----------
  for (int t = 0; t < 15; ++t) {
    // 1) fetch 16 patches as an 8x128-dword tile (row pitch 640) into sXin
    {
      const float* xr = x + (((size_t)(bImg * 15 + t)) * 480 + (size_t)py * 8) * 640 + px0 * 8;
#if HAVE_TDM
      if (wave == 0) {
        tdm_load_2d(lds_off(sXin), xr, /*tile*/ 128u, 8u, /*tensor*/ 640u, 480u, 640u);
        __builtin_amdgcn_s_wait_tensorcnt(0);
      }
#else
      #pragma unroll
      for (int i = 0; i < 8; ++i)
        sXin[i * 128 + tid] = xr[i * 640 + tid];
#endif
    }
    __syncthreads();
    // 2) embed: xe = xp @ W_embed^T + b_embed   (wave w -> N-tile w)
    {
      int n0 = wave * 16;
      v8f acc = wmma64_xin(sXin, sW + OFF_WE, n0, lane);
      float bias = b_embed[n0 + n];
      #pragma unroll
      for (int v = 0; v < 8; ++v) sX[(v + mo) * 64 + n0 + n] = acc[v] + bias;
    }
    __syncthreads();
    // 3) LN1 -> fx (bf16; keep f32 copy at t=14 for residuals)
    layernorm_16x64(sX, g1, be1, sFxB, (t == 14) ? sFx14 : nullptr, sMu, sRs, tid);
    // 4) Bu = fx @ Bbar^T  (cols 0..31 re, 32..63 im)
    {
      int n0 = wave * 16;
      v8f acc = wmma64(sFxB, sW + OFF_BB, n0, lane);
      #pragma unroll
      for (int v = 0; v < 8; ++v) sBu[(v + mo) * 64 + n0 + n] = acc[v];
    }
    __syncthreads();
    // 5) scan update: s = Lbar*s + Bu[t]   (complex, registers)
    #pragma unroll
    for (int q = 0; q < 4; ++q) {
      int e = tid * 4 + q, si = e >> 5, p = e & 31;
      float br = sBu[si * 64 + p];
      float bi = sBu[si * 64 + 32 + p];
      float nr = l_re[q] * s_re[q] - l_im[q] * s_im[q] + br;
      float ni = l_re[q] * s_im[q] + l_im[q] * s_re[q] + bi;
      s_re[q] = nr; s_im[q] = ni;
    }
  }

  // ---------- last-step head (t = 14 only) ----------
  // 6) xs -> bf16 A tile [re | im]
  #pragma unroll
  for (int q = 0; q < 4; ++q) {
    int e = tid * 4 + q, si = e >> 5, p = e & 31;
    sA[si * 64 + p]      = f2bf(s_re[q]);
    sA[si * 64 + 32 + p] = f2bf(s_im[q]);
  }
  __syncthreads();
  // 7) ys = xs @ Ccat^T + fx*D ; xb = gelu(ys) + fx
  {
    int n0 = wave * 16;
    v8f acc = wmma64(sA, sW + OFF_CC, n0, lane);
    float dcol = Dv[n0 + n];
    #pragma unroll
    for (int v = 0; v < 8; ++v) {
      float fx = sFx14[(v + mo) * 64 + n0 + n];
      float ys = acc[v] + fx * dcol;
      sX[(v + mo) * 64 + n0 + n] = gelu_exact(ys) + fx;
    }
  }
  __syncthreads();
  // 8) LN2 -> fx2 (bf16 + f32)
  layernorm_16x64(sX, g2, be2, sFxB, sFx2f, sMu, sRs, tid);
  // 9) enc = fx2 @ W_enc^T (16x128): each wave 2 N-tiles
  {
    #pragma unroll
    for (int tt = 0; tt < 2; ++tt) {
      int n0 = (wave + tt * 4) * 16;
      v8f acc = wmma64(sFxB, sW + OFF_WENC, n0, lane);
      #pragma unroll
      for (int v = 0; v < 8; ++v) sEnc[(v + mo) * 128 + n0 + n] = acc[v];
    }
  }
  __syncthreads();
  // 10) GLU: h = a * gelu(gate)
  #pragma unroll
  for (int e = tid; e < 1024; e += NTHR) {
    int r = e >> 6, k = e & 63;
    float a  = sEnc[r * 128 + k];
    float gt = sEnc[r * 128 + 64 + k];
    sA[e] = f2bf(a * gelu_exact(gt));
  }
  __syncthreads();
  // 11) dec: xl = h @ W_dec^T + fx2
  {
    int n0 = wave * 16;
    v8f acc = wmma64(sA, sW + OFF_WDEC, n0, lane);
    #pragma unroll
    for (int v = 0; v < 8; ++v)
      sX[(v + mo) * 64 + n0 + n] = acc[v] + sFx2f[(v + mo) * 64 + n0 + n];
  }
  __syncthreads();
  // 12) LN3 -> xlast (bf16)
  layernorm_16x64(sX, g3, be3, sFxB, nullptr, sMu, sRs, tid);
  // 13) out: proj = xlast @ W_out^T + b_out, per output-channel chunk (64 cols)
  for (int oc = 0; oc < 16; ++oc) {
    int nt = oc * 4 + wave;                       // N-tile 0..63
    v8f acc = wmma64(sFxB, wWout, nt * 16, lane); // B streamed from global bf16
    #pragma unroll
    for (int v = 0; v < 8; ++v) sOutT[(v + mo) * 64 + wave * 16 + n] = acc[v];
    __syncthreads();
    {
      float* yr = y + (((size_t)(bImg * 16 + oc)) * 480 + (size_t)py * 8) * 640 + px0 * 8;
      #pragma unroll
      for (int i = 0; i < 8; ++i)
        yr[i * 640 + tid] = sOutT[s_ * 64 + i * 8 + j_] + b_out[oc * 64 + i * 8 + j_];
    }
    __syncthreads();
  }
}

extern "C" void kernel_launch(void* const* d_in, const int* in_sizes, int n_in,
                              void* d_out, int out_size, void* d_ws, size_t ws_size,
                              hipStream_t stream) {
  (void)in_sizes; (void)n_in; (void)out_size; (void)ws_size;
  const float* x        = (const float*)d_in[0];
  const float* W_embed  = (const float*)d_in[1];
  const float* b_embed  = (const float*)d_in[2];
  const float* g1       = (const float*)d_in[3];
  const float* be1      = (const float*)d_in[4];
  const float* Lre      = (const float*)d_in[5];
  const float* Lim      = (const float*)d_in[6];
  const float* B_re     = (const float*)d_in[7];
  const float* B_im     = (const float*)d_in[8];
  const float* C_re     = (const float*)d_in[9];
  const float* C_im     = (const float*)d_in[10];
  const float* Dv       = (const float*)d_in[11];
  const float* log_step = (const float*)d_in[12];
  const float* g2       = (const float*)d_in[13];
  const float* be2      = (const float*)d_in[14];
  const float* W_enc    = (const float*)d_in[15];
  const float* W_dec    = (const float*)d_in[16];
  const float* g3       = (const float*)d_in[17];
  const float* be3      = (const float*)d_in[18];
  const float* W_out    = (const float*)d_in[19];
  const float* b_out    = (const float*)d_in[20];

  char* ws = (char*)d_ws;
  bf16_t* wWb   = (bf16_t*)(ws + WS_WB);
  float*  wLbar = (float*)(ws + WS_LBAR);
  bf16_t* wWout = (bf16_t*)(ws + WS_WOUT);

  ssm_prep<<<96, 256, 0, stream>>>(W_embed, Lre, Lim, B_re, B_im, C_re, C_im,
                                   log_step, W_enc, W_dec, W_out,
                                   wWb, wLbar, wWout);
  ssm_fused<<<NBLK, NTHR, 0, stream>>>(x, b_embed, g1, be1, Dv, g2, be2, g3, be3,
                                       b_out, wWb, wLbar, wWout, (float*)d_out);
}
